// JEPAByteModel_30451318129184
// MI455X (gfx1250) — matrix operand order repaired and checked by reading, not compiled, not run on hardware
//
#include <hip/hip_runtime.h>
#include <hip/hip_bf16.h>

// ---------------- model constants ----------------
#define BB 4
#define SS 4096
#define DD 1024
#define HH 16
#define HD 64
#define MC 1024
#define LL 4
#define DF 4096         // MLP_MULT * D
#define RMS_EPS 1.1920929e-7f
#define LN10000 9.210340371976184f

typedef __attribute__((ext_vector_type(16))) __bf16 v16bf;
typedef __attribute__((ext_vector_type(4)))  __bf16 v4bf;
typedef __attribute__((ext_vector_type(2)))  __bf16 v2bf;
typedef __attribute__((ext_vector_type(8)))  float  v8f;

static __device__ __forceinline__ v4bf cvt4(const float4 f) {
  v4bf o;
  o[0] = (__bf16)f.x; o[1] = (__bf16)f.y; o[2] = (__bf16)f.z; o[3] = (__bf16)f.w;
  return o;
}

// =====================================================================
// 1) chunk ids: per-batch serial cumsum (tiny work)
// =====================================================================
__global__ void k_ids(const int* __restrict__ bnd, int* __restrict__ ids,
                      int* __restrict__ nch) {
  int b = blockIdx.x;
  if (threadIdx.x == 0) {
    int acc = 0;
    for (int s = 0; s < SS; ++s) {
      acc += bnd[b * SS + s];
      ids[b * SS + s] = acc - 1;
    }
    nch[b] = acc;
  }
}

// chunk start positions
__global__ void k_cstart(const int* __restrict__ ids, int* __restrict__ cstart) {
  int t = blockIdx.x * blockDim.x + threadIdx.x;
  if (t >= BB * SS) return;
  int b = t / SS, s = t % SS;
  int id = ids[t];
  if (s == 0 || ids[t - 1] != id) cstart[b * MC + id] = s;
}

// =====================================================================
// 2) chunk mean -> latents (writes h and x0); invalid chunks -> 0
// =====================================================================
__global__ void k_chunk_mean(const float* __restrict__ x,
                             const int* __restrict__ cstart,
                             const int* __restrict__ nch,
                             float* __restrict__ h, float* __restrict__ x0) {
  int c = blockIdx.x, b = blockIdx.y, tid = threadIdx.x;
  int nc = nch[b];
  size_t obase = ((size_t)b * MC + c) * DD;
  if (c >= nc) {
    for (int d = tid; d < DD; d += 256) { h[obase + d] = 0.f; x0[obase + d] = 0.f; }
    return;
  }
  int s0 = cstart[b * MC + c];
  int s1 = (c + 1 < nc) ? cstart[b * MC + c + 1] : SS;
  float inv = 1.f / (float)(s1 - s0);
  for (int d = tid; d < DD; d += 256) {
    float acc = 0.f;
    for (int s = s0; s < s1; ++s) acc += x[((size_t)b * SS + s) * DD + d];
    float m = acc * inv;
    h[obase + d] = m; x0[obase + d] = m;
  }
}

// =====================================================================
// elementwise: h = mix0[d]*h + mix1[d]*x0
// =====================================================================
__global__ void k_mix(float* __restrict__ h, const float* __restrict__ x0,
                      const float* __restrict__ mix) {
  size_t i = (size_t)blockIdx.x * blockDim.x + threadIdx.x;
  size_t n = (size_t)BB * MC * DD;
  if (i < n) {
    int d = (int)(i % DD);
    h[i] = mix[d] * h[i] + mix[DD + d] * x0[i];
  }
}

// =====================================================================
// row-wise RMS over D=1024
// =====================================================================
__global__ void k_rmsrow(const float* __restrict__ X, float* __restrict__ Y) {
  int row = blockIdx.x, tid = threadIdx.x;
  const float* p = X + (size_t)row * DD;
  float* q = Y + (size_t)row * DD;
  __shared__ float sh[256];
  float ss = 0.f;
  for (int d = tid; d < DD; d += 256) { float v = p[d]; ss += v * v; }
  sh[tid] = ss; __syncthreads();
  for (int o = 128; o > 0; o >>= 1) { if (tid < o) sh[tid] += sh[tid + o]; __syncthreads(); }
  float r = rsqrtf(sh[0] * (1.f / DD) + RMS_EPS);
  for (int d = tid; d < DD; d += 256) q[d] = p[d] * r;
}

// =====================================================================
// per-head RMS + RoPE (+optional gain). One block per (b,c,h), 64 threads.
// =====================================================================
__global__ void k_ropenorm(float* __restrict__ X, const float* __restrict__ gain,
                           int useGain) {
  int idx = blockIdx.x;                    // (b*MC + c)*H + h
  int hh = idx % HH;
  int c = (idx / HH) % MC;
  int tid = threadIdx.x;                   // 64
  float* p = X + (size_t)idx * HD;
  __shared__ float sh[64];
  float v = p[tid];
  sh[tid] = v * v; __syncthreads();
  for (int o = 32; o > 0; o >>= 1) { if (tid < o) sh[tid] += sh[tid + o]; __syncthreads(); }
  float ss = sh[0]; __syncthreads();
  float r = v * rsqrtf(ss * (1.f / 64.f) + RMS_EPS);
  sh[tid] = r; __syncthreads();
  if (tid < 32) {
    float fr = (float)c * __expf(-((float)tid) * (1.f / 32.f) * LN10000);
    float cs = cosf(fr), sn = sinf(fr);
    float g = useGain ? gain[hh] : 1.f;
    float x1 = sh[tid], x2 = sh[tid + 32];
    p[tid]      = (x1 * cs + x2 * sn) * g;
    p[tid + 32] = (-x1 * sn + x2 * cs) * g;
  }
}

// =====================================================================
// WMMA bf16 GEMM with ASYNC global->LDS double-buffered pipeline:
//   Y[M,N] = epilogue( X[M,K] @ W[N,K]^T )
//   mode 0: Y = acc   mode 1: Y = relu(acc)^2   mode 2: Y = R + cs[n]*acc
// Block: 128 threads = 4 waves; tile 64(M) x 64(N); k-step 32.
// Stage A: global_load_async_to_lds_b128 f32 tiles (ASYNCcnt pipelined)
// Stage B: LDS f32 -> LDS bf16 (packed cvt, b128 read / b64 store)
// Stage C: ds_load_b128 fragments + v_wmma_f32_16x16x32_bf16
// =====================================================================
#define GS 40    // bf16 LDS row stride: 80B, 16B-aligned subchunks
#define GF 36    // f32 staging row stride: 144B, 16B-aligned subchunks
__global__ __launch_bounds__(128)
void k_gemm(const float* __restrict__ X, const float* __restrict__ W,
            float* __restrict__ Y, const float* __restrict__ R,
            const float* __restrict__ cs, int M, int N, int K, int mode) {
  __shared__ float Af[2][64][GF];
  __shared__ float Bf[2][64][GF];
  __shared__ __bf16 As[64][GS];
  __shared__ __bf16 Bs[64][GS];
  const int tid = threadIdx.x;
  const int lane = tid & 31;
  const int wave = tid >> 5;
  const int bm = blockIdx.x * 64;
  const int bn = blockIdx.y * 64;

  v8f acc[4];
#pragma unroll
  for (int j = 0; j < 4; ++j)
#pragma unroll
    for (int r = 0; r < 8; ++r) acc[j][r] = 0.f;

  // issue one K-tile's async copies: 512 chunks of 16B per matrix,
  // 4 insts/thread/matrix -> 8 outstanding per wave per tile.
  auto issue = [&](int buf, int k0) {
#pragma unroll
    for (int inst = 0; inst < 4; ++inst) {
      int i = tid + inst * 128;            // 0..511
      int r = i >> 3, c4 = (i & 7) << 2;   // row, f32 col (16B chunks)
      unsigned long long gx =
          (unsigned long long)(size_t)&X[(size_t)(bm + r) * K + k0 + c4];
      unsigned long long gw =
          (unsigned long long)(size_t)&W[(size_t)(bn + r) * K + k0 + c4];
      unsigned lx = (unsigned)(size_t)&Af[buf][r][c4];
      unsigned lw = (unsigned)(size_t)&Bf[buf][r][c4];
      asm volatile("global_load_async_to_lds_b128 %0, %1, off"
                   :: "v"(lx), "v"(gx) : "memory");
      asm volatile("global_load_async_to_lds_b128 %0, %1, off"
                   :: "v"(lw), "v"(gw) : "memory");
    }
  };

  const int T = K >> 5;
  issue(0, 0);
  for (int t = 0; t < T; ++t) {
    const int buf = t & 1;
    if (t + 1 < T) {
      issue(buf ^ 1, (t + 1) << 5);
      asm volatile("s_wait_asynccnt 0x8" ::: "memory");   // tile t landed
    } else {
      asm volatile("s_wait_asynccnt 0x0" ::: "memory");   // drain
    }
    __syncthreads();

    // convert f32 staging -> bf16 compute tiles (4 elems/thread/step)
    for (int i = tid; i < 64 * 8; i += 128) {
      int r = i >> 3, c4 = (i & 7) << 2;
      float4 xa = *(const float4*)&Af[buf][r][c4];
      float4 xb = *(const float4*)&Bf[buf][r][c4];
      *(v4bf*)&As[r][c4] = cvt4(xa);
      *(v4bf*)&Bs[r][c4] = cvt4(xb);
    }
    __syncthreads();

    // A fragment (rows wave*16 .. +15, K=0..31)
    v16bf a;
    {
      int row = (wave << 4) + (lane & 15);
      int kh = lane >> 4;
#pragma unroll
      for (int i = 0; i < 16; ++i) {
        int kk = (kh << 4) + i;
        a[i] = As[row][kk];
      }
    }
#pragma unroll
    for (int j = 0; j < 4; ++j) {
      v16bf bf;
      int col = (j << 4) + (lane & 15);
      int kb = (lane >> 4) << 4;
#pragma unroll
      for (int i = 0; i < 16; ++i) bf[i] = Bs[col][kb + i];
      acc[j] = __builtin_amdgcn_wmma_f32_16x16x32_bf16(
          false, a, false, bf, (short)0, acc[j], false, false);
    }
    __syncthreads();
  }

  const int halfadd = (lane >= 16) ? 8 : 0;
  const int coln = lane & 15;
#pragma unroll
  for (int j = 0; j < 4; ++j) {
    int col = bn + (j << 4) + coln;
#pragma unroll
    for (int r = 0; r < 8; ++r) {
      int row = bm + (wave << 4) + r + halfadd;
      size_t idx = (size_t)row * N + col;
      float v = acc[j][r];
      if (mode == 0) Y[idx] = v;
      else if (mode == 1) { float t = fmaxf(v, 0.f); Y[idx] = t * t; }
      else Y[idx] = R[idx] + cs[col] * v;
    }
  }
}

// =====================================================================
// Flash attention: per (b,h), q-tile of 64 rows, stream causal K/V tiles.
// Q,K,V,Y layout: [(b*MC + t)*H + h]*HD + d  (i.e. (B*C, D) row-major)
// V is stored TRANSPOSED in LDS (vt[d][t]) so PV B-fragments read
// contiguous runs (ds_load_b128) like the K fragments do.
// =====================================================================
#define FS 72  // LDS row stride (bf16): 144B rows, 16B-aligned
__global__ __launch_bounds__(128)
void k_flash(const float* __restrict__ Q, const float* __restrict__ Kb,
             const float* __restrict__ Vb, float* __restrict__ Y) {
  __shared__ __bf16 qs[64][FS];
  __shared__ __bf16 ks[64][FS];
  __shared__ __bf16 vt[64][FS];   // transposed: vt[d][t]
  __shared__ __bf16 ps[64][FS];
  __shared__ float sc[64][65];
  __shared__ float mrow[64], lrow[64], srow[64];
  const int tid = threadIdx.x, lane = tid & 31, wave = tid >> 5;
  const int qt = blockIdx.x;
  const int bh = blockIdx.y;
  const int b = bh / HH, hh = bh % HH;

  for (int i = tid; i < 64 * 16; i += 128) {
    int r = i >> 4, c4 = (i & 15) << 2;
    int t = qt * 64 + r;
    float4 qv = *(const float4*)&Q[((size_t)(b * MC + t) * HH + hh) * HD + c4];
    *(v4bf*)&qs[r][c4] = cvt4(qv);
  }
  if (tid < 64) { mrow[tid] = -3.0e38f; lrow[tid] = 0.f; }

  v8f o[4];
#pragma unroll
  for (int j = 0; j < 4; ++j)
#pragma unroll
    for (int r = 0; r < 8; ++r) o[j][r] = 0.f;

  for (int kt = 0; kt <= qt; ++kt) {
    __syncthreads();
    for (int i = tid; i < 64 * 16; i += 128) {
      int r = i >> 4, c4 = (i & 15) << 2;
      int t = kt * 64 + r;
      size_t idx = ((size_t)(b * MC + t) * HH + hh) * HD + c4;
      float4 kv = *(const float4*)&Kb[idx];
      float4 vv = *(const float4*)&Vb[idx];
      *(v4bf*)&ks[r][c4] = cvt4(kv);
      vt[c4 + 0][r] = (__bf16)vv.x;   // transpose V on store
      vt[c4 + 1][r] = (__bf16)vv.y;
      vt[c4 + 2][r] = (__bf16)vv.z;
      vt[c4 + 3][r] = (__bf16)vv.w;
    }
    __syncthreads();

    // scores strip: S = q @ k^T (K=64, two chained wmma), scale+mask -> sc
    {
      int arow = (wave << 4) + (lane & 15);
      int kh = lane >> 4;
      v16bf a0, a1;
#pragma unroll
      for (int i = 0; i < 16; ++i) {
        int kk = (kh << 4) + i;
        a0[i] = qs[arow][kk];
        a1[i] = qs[arow][32 + kk];
      }
      const int halfadd = (lane >= 16) ? 8 : 0;
#pragma unroll
      for (int j = 0; j < 4; ++j) {
        v16bf b0, b1;
        int col = (j << 4) + (lane & 15);
        int kb = (lane >> 4) << 4;
#pragma unroll
        for (int i = 0; i < 16; ++i) { b0[i] = ks[col][kb + i]; b1[i] = ks[col][32 + kb + i]; }
        v8f s;
#pragma unroll
        for (int r = 0; r < 8; ++r) s[r] = 0.f;
        s = __builtin_amdgcn_wmma_f32_16x16x32_bf16(false, a0, false, b0, (short)0, s, false, false);
        s = __builtin_amdgcn_wmma_f32_16x16x32_bf16(false, a1, false, b1, (short)0, s, false, false);
#pragma unroll
        for (int r = 0; r < 8; ++r) {
          int m = (wave << 4) + r + halfadd;
          int n = (j << 4) + (lane & 15);
          int grow = qt * 64 + m, gcol = kt * 64 + n;
          sc[m][n] = (gcol <= grow) ? s[r] * 0.125f : -3.0e38f;
        }
      }
    }
    __syncthreads();

    // online softmax per row (paired bf16 stores)
    if (tid < 64) {
      int m = tid;
      float mx = -3.0e38f;
      for (int n = 0; n < 64; ++n) mx = fmaxf(mx, sc[m][n]);
      float mold = mrow[m];
      float mnew = fmaxf(mold, mx);
      float corr = __expf(mold - mnew);
      float sum = 0.f;
      for (int n = 0; n < 64; n += 2) {
        float p0 = __expf(sc[m][n] - mnew);
        float p1 = __expf(sc[m][n + 1] - mnew);
        v2bf pp; pp[0] = (__bf16)p0; pp[1] = (__bf16)p1;
        *(v2bf*)&ps[m][n] = pp;
        sum += p0 + p1;
      }
      lrow[m] = lrow[m] * corr + sum;
      mrow[m] = mnew;
      srow[m] = corr;
    }
    __syncthreads();

    // O = O*corr + P @ V   (V fragments from transposed tile: contiguous)
    {
      const int halfadd = (lane >= 16) ? 8 : 0;
      int arow = (wave << 4) + (lane & 15);
      int kh = lane >> 4;
      v16bf a0, a1;
#pragma unroll
      for (int i = 0; i < 16; ++i) {
        int kk = (kh << 4) + i;
        a0[i] = ps[arow][kk];
        a1[i] = ps[arow][32 + kk];
      }
#pragma unroll
      for (int j = 0; j < 4; ++j) {
#pragma unroll
        for (int r = 0; r < 8; ++r) {
          int m = (wave << 4) + r + halfadd;
          o[j][r] *= srow[m];
        }
        v16bf b0, b1;
        int col = (j << 4) + (lane & 15);
        int kb = (lane >> 4) << 4;
#pragma unroll
        for (int i = 0; i < 16; ++i) { b0[i] = vt[col][kb + i]; b1[i] = vt[col][32 + kb + i]; }
        o[j] = __builtin_amdgcn_wmma_f32_16x16x32_bf16(false, a0, false, b0, (short)0, o[j], false, false);
        o[j] = __builtin_amdgcn_wmma_f32_16x16x32_bf16(false, a1, false, b1, (short)0, o[j], false, false);
      }
    }
  }
  __syncthreads();
  {
    const int halfadd = (lane >= 16) ? 8 : 0;
#pragma unroll
    for (int j = 0; j < 4; ++j) {
      int c = (j << 4) + (lane & 15);
#pragma unroll
      for (int r = 0; r < 8; ++r) {
        int m = (wave << 4) + r + halfadd;
        int t = qt * 64 + m;
        Y[((size_t)(b * MC + t) * HH + hh) * HD + c] = o[j][r] / lrow[m];
      }
    }
  }
}

// =====================================================================
// final valid mask -> d_out
// =====================================================================
__global__ void k_final(const float* __restrict__ h, const int* __restrict__ nch,
                        float* __restrict__ out) {
  size_t i = (size_t)blockIdx.x * blockDim.x + threadIdx.x;
  size_t n = (size_t)BB * MC * DD;
  if (i < n) {
    size_t bc = i / DD;
    int c = (int)(bc % MC);
    int b = (int)(bc / MC);
    out[i] = (c < nch[b]) ? h[i] : 0.f;
  }
}

// =====================================================================
// host orchestration
// =====================================================================
extern "C" void kernel_launch(void* const* d_in, const int* in_sizes, int n_in,
                              void* d_out, int out_size, void* d_ws, size_t ws_size,
                              hipStream_t stream) {
  (void)in_sizes; (void)n_in; (void)out_size; (void)ws_size;
  const float* x    = (const float*)d_in[0];
  const int*   bnd  = (const int*)d_in[1];
  const float* Wq   = (const float*)d_in[2];
  const float* Wk   = (const float*)d_in[3];
  const float* Wv   = (const float*)d_in[4];
  const float* Wo   = (const float*)d_in[5];
  const float* qg   = (const float*)d_in[6];
  const float* fc   = (const float*)d_in[7];
  const float* proj = (const float*)d_in[8];
  const float* ascl = (const float*)d_in[9];
  const float* mscl = (const float*)d_in[10];
  const float* rmix = (const float*)d_in[11];
  float* out = (float*)d_out;

  char* ws = (char*)d_ws;
  size_t off = 0;
  auto alloc = [&](size_t bytes) -> void* {
    void* p = ws + off;
    off += (bytes + 255) & ~(size_t)255;
    return p;
  };
  const size_t NE = (size_t)BB * MC * DD;           // 4M activations
  int*   ids    = (int*)alloc((size_t)BB * SS * 4);
  int*   nch    = (int*)alloc(BB * 4);
  int*   cstart = (int*)alloc((size_t)BB * MC * 4);
  float* h      = (float*)alloc(NE * 4);
  float* x0     = (float*)alloc(NE * 4);
  float* xn     = (float*)alloc(NE * 4);
  float* qb     = (float*)alloc(NE * 4);
  float* kb     = (float*)alloc(NE * 4);
  float* vb     = (float*)alloc(NE * 4);
  float* yb     = (float*)alloc(NE * 4);
  float* hid    = (float*)alloc((size_t)BB * MC * DF * 4);

  const int M = BB * MC;                            // 4096 rows
  const int EB = (int)((NE + 255) / 256);

  k_ids<<<BB, 32, 0, stream>>>(bnd, ids, nch);
  k_cstart<<<(BB * SS + 255) / 256, 256, 0, stream>>>(ids, cstart);
  k_chunk_mean<<<dim3(MC, BB), 256, 0, stream>>>(x, cstart, nch, h, x0);

  for (int l = 0; l < LL; ++l) {
    k_mix<<<EB, 256, 0, stream>>>(h, x0, rmix + (size_t)l * 2 * DD);
    k_rmsrow<<<M, 256, 0, stream>>>(h, xn);
    k_gemm<<<dim3(M / 64, DD / 64), 128, 0, stream>>>(xn, Wq + (size_t)l * DD * DD, qb, nullptr, nullptr, M, DD, DD, 0);
    k_gemm<<<dim3(M / 64, DD / 64), 128, 0, stream>>>(xn, Wk + (size_t)l * DD * DD, kb, nullptr, nullptr, M, DD, DD, 0);
    k_gemm<<<dim3(M / 64, DD / 64), 128, 0, stream>>>(xn, Wv + (size_t)l * DD * DD, vb, nullptr, nullptr, M, DD, DD, 0);
    k_ropenorm<<<M * HH, 64, 0, stream>>>(qb, qg + (size_t)l * HH, 1);
    k_ropenorm<<<M * HH, 64, 0, stream>>>(kb, nullptr, 0);
    k_flash<<<dim3(MC / 64, BB * HH), 128, 0, stream>>>(qb, kb, vb, yb);
    k_gemm<<<dim3(M / 64, DD / 64), 128, 0, stream>>>(yb, Wo + (size_t)l * DD * DD, h, h, ascl + (size_t)l * DD, M, DD, DD, 2);
    k_rmsrow<<<M, 256, 0, stream>>>(h, xn);
    k_gemm<<<dim3(M / 64, DF / 64), 128, 0, stream>>>(xn, fc + (size_t)l * DF * DD, hid, nullptr, nullptr, M, DF, DD, 1);
    k_gemm<<<dim3(M / 64, DD / 64), 128, 0, stream>>>(hid, proj + (size_t)l * DD * DF, h, h, mscl + (size_t)l * DD, M, DD, DF, 2);
  }
  k_final<<<EB, 256, 0, stream>>>(h, nch, out);
}
